// YoutubeSBC_91079076479334
// MI455X (gfx1250) — matrix operand order repaired and checked by reading, not compiled, not run on hardware
//
#include <hip/hip_runtime.h>
#include <hip/hip_bf16.h>

typedef _Float16 v16h __attribute__((ext_vector_type(16)));
typedef _Float16 v8h  __attribute__((ext_vector_type(8)));
typedef _Float16 v4h  __attribute__((ext_vector_type(4)));
typedef float    v8f  __attribute__((ext_vector_type(8)));
typedef float    v4f  __attribute__((ext_vector_type(4)));
typedef int      v4i  __attribute__((ext_vector_type(4)));

union V16H { v16h v; v8h h[2]; };

#if __has_builtin(__builtin_amdgcn_global_load_async_to_lds_b128)
#define HAVE_ASYNC_LDS 1
#else
#define HAVE_ASYNC_LDS 0
#endif

#if HAVE_ASYNC_LDS
// AMDGPU address-space-qualified pointer types (same pattern HIP's own
// headers use for the *_global_* builtins): AS1 = global, AS3 = LDS.
typedef __attribute__((address_space(1))) v4i glb_v4i;
typedef __attribute__((address_space(3))) v4i lds_v4i;

__device__ __forceinline__ void async_gather_b128(const float* gsrc, float* ldst) {
  __builtin_amdgcn_global_load_async_to_lds_b128(
      (glb_v4i*)const_cast<float*>(gsrc),
      (lds_v4i*)ldst,
      0, 0);
}

__device__ __forceinline__ void async_wait_zero() {
#if __has_builtin(__builtin_amdgcn_s_wait_asynccnt)
  __builtin_amdgcn_s_wait_asynccnt(0);
#else
  asm volatile("s_wait_asynccnt 0" ::: "memory");
#endif
}
#endif

// ---------------------------------------------------------------------------
// Stage 0: weight transpose + f32->f16 convert.  in: [K][N] f32 row-major,
// out: [N][K] f16 so a WMMA B-fragment (two 8-K runs, 16 apart, at fixed N) is two
// contiguous 16B loads.
// ---------------------------------------------------------------------------
__global__ __launch_bounds__(256) void wconvert_kernel(
    const float* __restrict__ in, _Float16* __restrict__ out, int K, int N) {
  int idx = blockIdx.x * 256 + threadIdx.x;
  if (idx < K * N) {
    int k = idx / N, n = idx - k * N;
    out[n * K + k] = (_Float16)in[idx];
  }
}

// ---------------------------------------------------------------------------
// Stage 1: fused tower (gather -> GEMM1+bias+relu -> GEMM2+bias+relu).
// 64 batch rows per workgroup, 256 threads = 8 wave32s, one 16x16 output tile
// per wave per iteration.  All matrix math via v_wmma_f32_16x16x32_f16.
// Embedding gather uses GLOBAL_LOAD_ASYNC_TO_LDS_B128 (ASYNCcnt) when the
// toolchain exposes it: each lane fires 16B gathers straight into LDS with
// no VGPR landing zone, then an LDS->LDS pass converts f32 -> f16.
// ---------------------------------------------------------------------------
template<int FIELDS, int K1>
__global__ __launch_bounds__(256) void tower_kernel(
    const int* __restrict__ ids, const float* __restrict__ table,
    const _Float16* __restrict__ W1t, const float* __restrict__ b1,
    const _Float16* __restrict__ W2t, const float* __restrict__ b2,
    float* __restrict__ out) {
  constexpr int ROWS = 64;   // batch rows per workgroup
  constexpr int N1   = 256;  // hidden dim (== K2)
  constexpr int K2   = 256;
  constexpr int N2   = 128;  // tower output dim
  constexpr int D    = 64;   // embedding dim per field

  __shared__ int s_ids[ROWS * FIELDS];
  __shared__ __align__(16) _Float16 sA[ROWS * K1];   // gathered embeddings, f16

#if HAVE_ASYNC_LDS
  // f32 gather staging overlapped with the hidden-activation buffer: the
  // staging is dead before sH is first written (both sides fenced by
  // __syncthreads()).  Peak LDS: 64K(sA) + 128K(scratch) + 2K(ids) = 194K.
  constexpr size_t SE_BYTES = (size_t)ROWS * K1 * sizeof(float);
  constexpr size_t SH_BYTES = (size_t)ROWS * K2 * sizeof(_Float16);
  __shared__ __align__(16) char sScratch[SE_BYTES > SH_BYTES ? SE_BYTES : SH_BYTES];
  float*    sE = (float*)sScratch;
  _Float16* sH = (_Float16*)sScratch;
#else
  __shared__ __align__(16) _Float16 sH[ROWS * K2];   // hidden activations, f16
#endif

  const int tid  = threadIdx.x;
  const int wave = tid >> 5;
  const int lane = tid & 31;
  const int lr   = lane & 15;   // N / M index within fragment
  const int lh   = lane >> 4;   // half-wave selector (K offset)
  const int row0 = blockIdx.x * ROWS;

  // ---- load ids ----
  for (int q = tid; q < ROWS * FIELDS; q += 256)
    s_ids[q] = ids[row0 * FIELDS + q];
  __syncthreads();

  // ---- embedding gather into LDS ----
  constexpr int C4 = K1 / 4;               // float4s per row
#if HAVE_ASYNC_LDS
  // Phase A: fire-and-forget 16B async gathers (global -> LDS f32 staging).
#pragma unroll 4
  for (int q = tid; q < ROWS * C4; q += 256) {
    int row = q / C4, c = q - row * C4;
    int f = c / (D / 4), w4 = c - f * (D / 4);
    int id = s_ids[row * FIELDS + f];
    async_gather_b128(table + (size_t)id * D + w4 * 4,
                      &sE[row * K1 + c * 4]);
  }
  async_wait_zero();
  __syncthreads();
  // Phase B: LDS f32 -> LDS f16 convert (ds_load_b128 + packed cvt + ds_store_b64).
#pragma unroll 4
  for (int q = tid; q < ROWS * C4; q += 256) {
    int row = q / C4, c = q - row * C4;
    v4f x = *(const v4f*)&sE[row * K1 + c * 4];
    v4h h;
    h[0] = (_Float16)x[0]; h[1] = (_Float16)x[1];
    h[2] = (_Float16)x[2]; h[3] = (_Float16)x[3];
    *(v4h*)&sA[row * K1 + c * 4] = h;
  }
#else
  for (int q = tid; q < ROWS * C4; q += 256) {
    int row = q / C4, c = q - row * C4;
    int f = c / (D / 4), w4 = c - f * (D / 4);
    int id = s_ids[row * FIELDS + f];
    v4f x = *(const v4f*)(table + (size_t)id * D + w4 * 4);
    v4h h;
    h[0] = (_Float16)x[0]; h[1] = (_Float16)x[1];
    h[2] = (_Float16)x[2]; h[3] = (_Float16)x[3];
    *(v4h*)&sA[row * K1 + c * 4] = h;
  }
#endif
  __syncthreads();

  // ---- GEMM1: [ROWS,K1] x [K1,N1] -> relu -> sH ----
  for (int t = wave; t < (ROWS / 16) * (N1 / 16); t += 8) {
    int mt = t & (ROWS / 16 - 1);
    int nt = t / (ROWS / 16);
    int mb = mt * 16, nb = nt * 16;
    v8f acc = {0.f, 0.f, 0.f, 0.f, 0.f, 0.f, 0.f, 0.f};
#pragma unroll 4
    for (int kk = 0; kk < K1 / 32; ++kk) {
      V16H a, b;
      const _Float16* ap = &sA[(mb + lr) * K1 + kk * 32 + lh * 8];
      a.h[0] = *(const v8h*)ap;            // K = kk*32 + lh*8 .. +7   (v0..3)
      a.h[1] = *(const v8h*)(ap + 16);     // K = kk*32 + 16 + lh*8 .. (v4..7)
      const _Float16* bp = &W1t[(nb + lr) * K1 + kk * 32 + lh * 8];
      b.h[0] = *(const v8h*)bp;            // K = kk*32 + lh*8 .. +7
      b.h[1] = *(const v8h*)(bp + 16);     // K = kk*32 + 16 + lh*8 .. +7
      acc = __builtin_amdgcn_wmma_f32_16x16x32_f16(
          false, a.v, false, b.v, (short)0, acc, false, false);
    }
    int n = nb + lr;
    float bias = b1[n];
#pragma unroll
    for (int v = 0; v < 8; ++v) {
      int m = mb + lh * 8 + v;
      float x = acc[v] + bias;
      x = x > 0.f ? x : 0.f;
      sH[m * K2 + n] = (_Float16)x;
    }
  }
  __syncthreads();

  // ---- GEMM2: [ROWS,K2] x [K2,N2] -> relu -> global out [B][N2] f32 ----
  for (int t = wave; t < (ROWS / 16) * (N2 / 16); t += 8) {
    int mt = t & (ROWS / 16 - 1);
    int nt = t / (ROWS / 16);
    int mb = mt * 16, nb = nt * 16;
    v8f acc = {0.f, 0.f, 0.f, 0.f, 0.f, 0.f, 0.f, 0.f};
#pragma unroll
    for (int kk = 0; kk < K2 / 32; ++kk) {
      V16H a, b;
      const _Float16* ap = &sH[(mb + lr) * K2 + kk * 32 + lh * 8];
      a.h[0] = *(const v8h*)ap;
      a.h[1] = *(const v8h*)(ap + 16);
      const _Float16* bp = &W2t[(nb + lr) * K2 + kk * 32 + lh * 8];
      b.h[0] = *(const v8h*)bp;
      b.h[1] = *(const v8h*)(bp + 16);
      acc = __builtin_amdgcn_wmma_f32_16x16x32_f16(
          false, a.v, false, b.v, (short)0, acc, false, false);
    }
    int n = nb + lr;
    float bias = b2[n];
#pragma unroll
    for (int v = 0; v < 8; ++v) {
      int row = row0 + mb + lh * 8 + v;
      float x = acc[v] + bias;
      x = x > 0.f ? x : 0.f;
      out[(size_t)row * N2 + n] = x;
    }
  }
}

// ---------------------------------------------------------------------------
// Stage 2: per-row cosine similarity + log-weight correction.
// One wave32 per row; 128 dims -> 4 floats/lane; shfl_xor reduction.
// ---------------------------------------------------------------------------
__global__ __launch_bounds__(256) void cosine_kernel(
    const float* __restrict__ u, const float* __restrict__ v,
    const float* __restrict__ w, float* __restrict__ y, int B) {
  int wave = threadIdx.x >> 5, lane = threadIdx.x & 31;
  int row = blockIdx.x * 8 + wave;
  if (row >= B) return;
  v4f u4 = *(const v4f*)(u + (size_t)row * 128 + lane * 4);
  v4f v4 = *(const v4f*)(v + (size_t)row * 128 + lane * 4);
  float dot = 0.f, un = 0.f, vn = 0.f;
#pragma unroll
  for (int i = 0; i < 4; ++i) {
    dot += u4[i] * v4[i];
    un  += u4[i] * u4[i];
    vn  += v4[i] * v4[i];
  }
#pragma unroll
  for (int off = 16; off > 0; off >>= 1) {
    dot += __shfl_xor(dot, off);
    un  += __shfl_xor(un,  off);
    vn  += __shfl_xor(vn,  off);
  }
  if (lane == 0) {
    float d = fmaxf(sqrtf(un), 1e-8f) * fmaxf(sqrtf(vn), 1e-8f);
    y[row] = dot / d - logf(w[row]);
  }
}

// ---------------------------------------------------------------------------
// Stage 3: wrap-around in-batch negative gather: out[i][j] = y[(i+j)%B]
// ---------------------------------------------------------------------------
__global__ __launch_bounds__(256) void scores_kernel(
    const float* __restrict__ y, float* __restrict__ out,
    int B, int per, int total) {
  int t = blockIdx.x * 256 + threadIdx.x;
  if (t < total) {
    int i = t / per, j = t - i * per;
    int k = i + j;
    if (k >= B) k -= B;
    out[t] = y[k];
  }
}

// ---------------------------------------------------------------------------
extern "C" void kernel_launch(void* const* d_in, const int* in_sizes, int n_in,
                              void* d_out, int out_size, void* d_ws, size_t ws_size,
                              hipStream_t stream) {
  const int*   user_ids = (const int*)d_in[0];
  const int*   item_ids = (const int*)d_in[1];
  const float* sw       = (const float*)d_in[2];
  const float* utab     = (const float*)d_in[3];
  const float* itab     = (const float*)d_in[4];
  const float* uW1      = (const float*)d_in[5];
  const float* ub1      = (const float*)d_in[6];
  const float* uW2      = (const float*)d_in[7];
  const float* ub2      = (const float*)d_in[8];
  const float* iW1      = (const float*)d_in[9];
  const float* ib1      = (const float*)d_in[10];
  const float* iW2      = (const float*)d_in[11];
  const float* ib2      = (const float*)d_in[12];

  const int B   = in_sizes[2];          // batch (sample_weight is [B])
  const int per = out_size / B;         // n_neg + 1

  // workspace layout
  char* ws = (char*)d_ws;
  float* u = (float*)ws;            ws += (size_t)B * 128 * sizeof(float);
  float* v = (float*)ws;            ws += (size_t)B * 128 * sizeof(float);
  float* y = (float*)ws;            ws += (size_t)B * sizeof(float);
  _Float16* uW1t = (_Float16*)ws;   ws += (size_t)512 * 256 * sizeof(_Float16);
  _Float16* uW2t = (_Float16*)ws;   ws += (size_t)256 * 128 * sizeof(_Float16);
  _Float16* iW1t = (_Float16*)ws;   ws += (size_t)256 * 256 * sizeof(_Float16);
  _Float16* iW2t = (_Float16*)ws;   ws += (size_t)256 * 128 * sizeof(_Float16);

  // Stage 0: weights -> transposed f16 (tiny; stays resident in L2)
  wconvert_kernel<<<(512 * 256 + 255) / 256, 256, 0, stream>>>(uW1, uW1t, 512, 256);
  wconvert_kernel<<<(256 * 128 + 255) / 256, 256, 0, stream>>>(uW2, uW2t, 256, 128);
  wconvert_kernel<<<(256 * 256 + 255) / 256, 256, 0, stream>>>(iW1, iW1t, 256, 256);
  wconvert_kernel<<<(256 * 128 + 255) / 256, 256, 0, stream>>>(iW2, iW2t, 256, 128);

  // Stage 1: towers (64 rows per workgroup)
  tower_kernel<8, 512><<<B / 64, 256, 0, stream>>>(user_ids, utab, uW1t, ub1, uW2t, ub2, u);
  tower_kernel<4, 256><<<B / 64, 256, 0, stream>>>(item_ids, itab, iW1t, ib1, iW2t, ib2, v);

  // Stage 2: cosine + bias correction
  cosine_kernel<<<(B + 7) / 8, 256, 0, stream>>>(u, v, sw, y, B);

  // Stage 3: score gather
  scores_kernel<<<(out_size + 255) / 256, 256, 0, stream>>>(y, (float*)d_out, B, per, out_size);
}
